// FFTConv1D_86157043958228
// MI455X (gfx1250) — matrix operand order repaired and checked by reading, not compiled, not run on hardware
//
#include <hip/hip_runtime.h>

// ---------------------------------------------------------------------------
// FFT-conv reference == causal conv1d:
//   y[b,s,f] = bias[f] + sum_{k<33} sum_{c<64} x[b,s-k,c] * W[k,c,f]
// Implicit GEMM on CDNA5 v_wmma_f32_16x16x32_bf16 with f32 -> (bf16 hi + lo)
// split (3 WMMAs per f32 product, ~2^-15 relative accuracy).
// All operands are pre-packed into the exact WMMA bit layout (consecutive
// K-pairs per 32-bit word) so fragment builds are pure ds_load_b128.
// W is pre-packed ONCE into d_ws by a helper kernel (in-kernel pack fallback
// if ws_size is too small), so the main loop's W staging is a pure b128 copy.
// B=32 S=4096 C=64 K=33 F=128.
// ---------------------------------------------------------------------------

typedef __attribute__((ext_vector_type(16))) __bf16   v16bf;
typedef __attribute__((ext_vector_type(8)))  float    v8f;
typedef __attribute__((ext_vector_type(4)))  unsigned v4u;

struct U32x8 { v4u lo, hi; };   // 32B: bit_cast source for v16bf

#define B_DIM   32
#define S_DIM   4096
#define C_DIM   64
#define K_DIM   33
#define F_DIM   128

#define BLOCK_M 128
#define XROWS   160        // BLOCK_M + K_DIM - 1
#define XSTR    36         // packed-x row stride in words (32 pairs + 4 pad)
#define NCHUNK  (K_DIM * 2)        // 66 (k, c0) chunks
#define CHUNK_W 4096               // words per packed W chunk (2048 hi + 2048 lo)

// Pack two f32 into one word of bf16 "hi" parts and one word of bf16 "lo"
// residuals (truncation split; hi reconstructs exactly, lo holds next 8 bits).
__device__ __forceinline__ void pack_pair(float x0, float x1,
                                          unsigned &hw, unsigned &lw) {
    unsigned u0 = __float_as_uint(x0), u1 = __float_as_uint(x1);
    unsigned h0 = u0 & 0xffff0000u,    h1 = u1 & 0xffff0000u;
    hw = h1 | (u0 >> 16);
    unsigned r0 = __float_as_uint(x0 - __uint_as_float(h0));
    unsigned r1 = __float_as_uint(x1 - __uint_as_float(h1));
    lw = (r1 & 0xffff0000u) | (r0 >> 16);
}

// ---------------------------------------------------------------------------
// One-time W pre-pack: d_ws chunk kc holds 2048 hi words then 2048 lo words,
// in LDS order: word = ((p>>2)*128 + f)*4 + (p&3), p = K-pair, f = filter.
// ---------------------------------------------------------------------------
__global__ __launch_bounds__(256)
void pack_w_kernel(const float* __restrict__ w, unsigned* __restrict__ ws) {
    int idx = blockIdx.x * 256 + threadIdx.x;     // 66 * 2048 = 135168 words
    if (idx >= NCHUNK * 2048) return;
    int kc   = idx >> 11;
    int word = idx & 2047;
    int g  = word >> 9;                           // p >> 2
    int f  = (word >> 2) & 127;
    int p  = g * 4 + (word & 3);                  // K pair 0..15
    int k  = kc >> 1;
    int c0 = (kc & 1) << 5;
    const float* wp = w + ((size_t)(k * C_DIM + c0 + 2 * p) * F_DIM + f);
    unsigned hw, lw;
    pack_pair(wp[0], wp[F_DIM], hw, lw);
    ws[(size_t)kc * CHUNK_W + word]        = hw;
    ws[(size_t)kc * CHUNK_W + 2048 + word] = lw;
}

// ---------------------------------------------------------------------------
// Main kernel. USE_WS=true: W chunks copied from pre-packed d_ws (pure b128
// copies). USE_WS=false: self-contained per-chunk f32 load + pack.
// ---------------------------------------------------------------------------
template <bool USE_WS>
__global__ __launch_bounds__(256)
void fftconv1d_wmma_kernel(const float* __restrict__ x,
                           const float* __restrict__ w,
                           const float* __restrict__ bias,
                           float* __restrict__ out,
                           const unsigned* __restrict__ ws) {
    // Packed input window, bit-ready for A fragments: word = K-pair (2 bf16).
    __shared__ __align__(16) unsigned xh[XROWS * XSTR];      // 23040 B
    __shared__ __align__(16) unsigned xl[XROWS * XSTR];      // 23040 B
    // Packed W chunk: [0..2047] hi, [2048..4095] lo, layout [g][f][4].
    __shared__ __align__(16) unsigned whl[CHUNK_W];          // 16384 B

    const int tid  = threadIdx.x;
    const int lane = tid & 31;
    const int wave = tid >> 5;
    const int mw   = wave & 3;          // M offset 32*mw
    const int nw   = wave >> 2;         // N offset 64*nw
    const int lm   = lane & 15;
    const int h    = lane >> 4;

    const int b  = blockIdx.x >> 5;
    const int s0 = (blockIdx.x & 31) * BLOCK_M;

    // ---- stage + pack input window x[b, s0-32 .. s0+127, :] ----------------
    #pragma unroll
    for (int j = 0; j < 20; ++j) {
        int idx = j * 256 + tid;                 // 5120 packed words
        int row = idx >> 5;                      // 0..159
        int pr  = idx & 31;                      // channel pair 0..31
        int s   = s0 - 32 + row;
        float2 v = make_float2(0.f, 0.f);
        if (s >= 0)
            v = *(const float2*)(x + ((size_t)(b * S_DIM + s) * C_DIM + pr * 2));
        unsigned hw, lw;
        pack_pair(v.x, v.y, hw, lw);
        xh[row * XSTR + pr] = hw;
        xl[row * XSTR + pr] = lw;
    }

    v8f acc[2][4];
    #pragma unroll
    for (int mt = 0; mt < 2; ++mt)
        #pragma unroll
        for (int nt = 0; nt < 4; ++nt)
            acc[mt][nt] = (v8f){0.f, 0.f, 0.f, 0.f, 0.f, 0.f, 0.f, 0.f};

    // Software-pipelined W prefetch for chunk 0.
    v4u   wreg[4];                                // USE_WS path: 16 words/thread
    float wr0[8], wr1[8];                         // fallback path
    if (USE_WS) {
        #pragma unroll
        for (int j = 0; j < 4; ++j)
            wreg[j] = *(const v4u*)(ws + (size_t)tid * 16 + j * 4);
    } else {
        #pragma unroll
        for (int j = 0; j < 8; ++j) {
            int idx = j * 256 + tid;
            int p   = idx >> 7;
            int f   = idx & 127;
            const float* wp = w + ((size_t)(2 * p) * F_DIM + f);
            wr0[j] = wp[0];
            wr1[j] = wp[F_DIM];
        }
    }

    // ---- reduction: 33 taps x 2 chunks of 32 channels ----------------------
    for (int kc = 0; kc < NCHUNK; ++kc) {
        __syncthreads();                         // prior chunk's frag reads done
        if (USE_WS) {
            #pragma unroll
            for (int j = 0; j < 4; ++j)
                *(v4u*)&whl[tid * 16 + j * 4] = wreg[j];
        } else {
            #pragma unroll
            for (int j = 0; j < 8; ++j) {
                int idx = j * 256 + tid;
                int p   = idx >> 7;
                int f   = idx & 127;
                unsigned hw, lw;
                pack_pair(wr0[j], wr1[j], hw, lw);
                int word = ((p >> 2) * F_DIM + f) * 4 + (p & 3);
                whl[word]        = hw;
                whl[2048 + word] = lw;
            }
        }
        __syncthreads();

        // prefetch next chunk's W (latency hidden behind WMMAs below)
        if (kc + 1 < NCHUNK) {
            if (USE_WS) {
                const unsigned* src = ws + (size_t)(kc + 1) * CHUNK_W + tid * 16;
                #pragma unroll
                for (int j = 0; j < 4; ++j)
                    wreg[j] = *(const v4u*)(src + j * 4);
            } else {
                int kn  = (kc + 1) >> 1;
                int c0n = ((kc + 1) & 1) << 5;
                #pragma unroll
                for (int j = 0; j < 8; ++j) {
                    int idx = j * 256 + tid;
                    int p   = idx >> 7;
                    int f   = idx & 127;
                    const float* wp = w + ((size_t)(kn * C_DIM + c0n + 2 * p) * F_DIM + f);
                    wr0[j] = wp[0];
                    wr1[j] = wp[F_DIM];
                }
            }
        }

        const int k  = kc >> 1;
        const int c0 = (kc & 1) << 5;

        // ---- A fragments: pure ds_load_b128, already in WMMA layout --------
        v16bf ahi[2], alo[2];
        #pragma unroll
        for (int mt = 0; mt < 2; ++mt) {
            int row  = 32 * mw + 16 * mt + lm + 32 - k;     // 0..159
            int base = row * XSTR + (c0 >> 1) + 4 * h;
            U32x8 t;
            t.lo = *(const v4u*)&xh[base];
            t.hi = *(const v4u*)&xh[base + 8];
            ahi[mt] = __builtin_bit_cast(v16bf, t);
            t.lo = *(const v4u*)&xl[base];
            t.hi = *(const v4u*)&xl[base + 8];
            alo[mt] = __builtin_bit_cast(v16bf, t);
        }

        // ---- B fragments + WMMAs ------------------------------------------
        #pragma unroll
        for (int nt = 0; nt < 4; ++nt) {
            int n     = 64 * nw + 16 * nt + lm;
            int bbase = (h * F_DIM + n) * 4;
            U32x8 t;
            t.lo = *(const v4u*)&whl[bbase];
            t.hi = *(const v4u*)&whl[bbase + 2 * F_DIM * 4];
            v16bf bhi = __builtin_bit_cast(v16bf, t);
            t.lo = *(const v4u*)&whl[2048 + bbase];
            t.hi = *(const v4u*)&whl[2048 + bbase + 2 * F_DIM * 4];
            v16bf blo = __builtin_bit_cast(v16bf, t);

            #pragma unroll
            for (int mt = 0; mt < 2; ++mt) {
                // hi*hi + lo*hi + hi*lo  (drop lo*lo, ~2^-16 relative)
                acc[mt][nt] = __builtin_amdgcn_wmma_f32_16x16x32_bf16(
                    false, ahi[mt], false, bhi, (short)0, acc[mt][nt], false, false);
                acc[mt][nt] = __builtin_amdgcn_wmma_f32_16x16x32_bf16(
                    false, alo[mt], false, bhi, (short)0, acc[mt][nt], false, false);
                acc[mt][nt] = __builtin_amdgcn_wmma_f32_16x16x32_bf16(
                    false, ahi[mt], false, blo, (short)0, acc[mt][nt], false, false);
            }
        }
    }

    // ---- store: D(f32 16x16): lane n = lane%16, rows m = 8h + r ------------
    #pragma unroll
    for (int mt = 0; mt < 2; ++mt) {
        #pragma unroll
        for (int nt = 0; nt < 4; ++nt) {
            int n    = 64 * nw + 16 * nt + lm;
            float bv = bias[n];
            #pragma unroll
            for (int r = 0; r < 8; ++r) {
                int m = 32 * mw + 16 * mt + 8 * h + r;
                out[((size_t)(b * S_DIM + s0 + m)) * F_DIM + n] = acc[mt][nt][r] + bv;
            }
        }
    }
}

extern "C" void kernel_launch(void* const* d_in, const int* in_sizes, int n_in,
                              void* d_out, int out_size, void* d_ws, size_t ws_size,
                              hipStream_t stream) {
    (void)in_sizes; (void)n_in; (void)out_size;
    const float* x    = (const float*)d_in[0];   // [32, 4096, 64]
    const float* w    = (const float*)d_in[1];   // [33, 64, 128]
    const float* bias = (const float*)d_in[2];   // [128]
    float* out        = (float*)d_out;           // [32, 4096, 128]

    dim3 grid(B_DIM * (S_DIM / BLOCK_M));        // 1024 workgroups
    dim3 block(256);                             // 8 wave32

    const size_t ws_need = (size_t)NCHUNK * CHUNK_W * sizeof(unsigned); // 1.06 MB
    if (ws_size >= ws_need) {
        unsigned* wpk = (unsigned*)d_ws;
        pack_w_kernel<<<dim3((NCHUNK * 2048 + 255) / 256), block, 0, stream>>>(w, wpk);
        fftconv1d_wmma_kernel<true><<<grid, block, 0, stream>>>(x, w, bias, out, wpk);
    } else {
        fftconv1d_wmma_kernel<false><<<grid, block, 0, stream>>>(x, w, bias, out, nullptr);
    }
}
